// _autoencoder_12910671692497
// MI455X (gfx1250) — matrix operand order repaired and checked by reading, not compiled, not run on hardware
//
#include <hip/hip_runtime.h>
#include <hip/hip_bf16.h>
#include <math.h>

// ---------------------------------------------------------------------------
// Sizes (match reference)
#define BROWS 16384
#define DDIM  1024
#define ADIM  64
#define HDIM  512
#define NEXP  5
#define CATD  1036     // 512 + 8 + 4 + 512
#define CATP  1088     // CATD padded up to a multiple of 64 (zero-filled)

typedef unsigned short u16;
typedef __attribute__((ext_vector_type(16))) __bf16        v16bf;
typedef __attribute__((ext_vector_type(8)))  float         v8f;
typedef __attribute__((ext_vector_type(4)))  unsigned int  u32x4;
typedef __attribute__((ext_vector_type(2)))  unsigned int  u32x2;

union Frag16 { v16bf bf; u32x4 q[2]; };

__device__ __forceinline__ u16 bfbits(float f) {
    unsigned u = __float_as_uint(f);
    return (u16)((u + 0x7FFFu + ((u >> 16) & 1u)) >> 16);   // RNE fp32 -> bf16
}
__device__ __forceinline__ unsigned pack2(float a, float b) {
    return (unsigned)bfbits(a) | ((unsigned)bfbits(b) << 16);
}

// CDNA5 async global->LDS copy (ISA ch.10.7 / 15.18), tracked by ASYNCcnt.
// Per-lane: LDS[ldsOff .. +15] = MEM[gaddr .. +15].
__device__ __forceinline__ void async_g2l_b128(const void* lds, const void* gaddr) {
    unsigned ldsOff = (unsigned)(unsigned long long)lds;    // low 32b of flat = LDS offset
    asm volatile("global_load_async_to_lds_b128 %0, %1, off"
                 :: "v"(ldsOff), "v"((unsigned long long)gaddr)
                 : "memory");
}
__device__ __forceinline__ void wait_async0() {
    asm volatile("s_wait_asynccnt 0" ::: "memory");
}

// ---------------------------------------------------------------------------
// Grouping kernels: bucket rows by expert label
__global__ void k_zero24(int* p) { if (threadIdx.x < 24) p[threadIdx.x] = 0; }

__global__ void k_count(const int* __restrict__ labels, int* __restrict__ cnts) {
    int b = blockIdx.x * blockDim.x + threadIdx.x;
    if (b < BROWS) atomicAdd(&cnts[labels[b]], 1);
}

__global__ void k_scan(const int* __restrict__ cnts, int* __restrict__ offs,
                       int* __restrict__ curs) {
    if (threadIdx.x == 0) {
        int s = 0;
        for (int e = 0; e < NEXP; ++e) { offs[e] = s; curs[e] = s; s += cnts[e]; }
        offs[NEXP] = s;
    }
}

__global__ void k_fill(const int* __restrict__ labels, int* __restrict__ curs,
                       int* __restrict__ idx) {
    int b = blockIdx.x * blockDim.x + threadIdx.x;
    if (b < BROWS) {
        int p = atomicAdd(&curs[labels[b]], 1);
        idx[p] = b;
    }
}

// ---------------------------------------------------------------------------
// Weight convert + transpose: W [K][N] fp32 -> WT [N][Kpad] bf16 (zero-padded).
// 2-D grid: x covers Kpad (64/block), y covers N.  No divisions.
__global__ void k_wconv(const float* __restrict__ W, u16* __restrict__ WT,
                        int K, int N, int Kpad)
{
    int k = blockIdx.x * 64 + threadIdx.x;
    int n = blockIdx.y;
    float v = (k < K) ? W[(long)k * N + n] : 0.f;
    WT[(long)n * Kpad + k] = bfbits(v);
}

// ---------------------------------------------------------------------------
// Grouped / dense bf16-WMMA GEMM.
//   Out[g, outOff+n] = act( A[g, :K] @ WT[n, :K]^T + bias[n] )
// A: fp32 (convert+pack path) or bf16 (async global->LDS path); WT pre-
// transposed/zero-padded bf16.  256 threads = 8 waves (4M x 2N); block tile
// 128x64; K-step 64 => 8 v_wmma per stage; branch-free inner loop (K is a
// multiple of 64, pad regions are zero).
template <bool AIN_F32, bool OUT_BF16>
__global__ __launch_bounds__(256)
void wmma_gemm(const void* __restrict__ Aptr, int ldA,
               const u16* __restrict__ WT, int ldW,
               const float* __restrict__ bias,
               void* __restrict__ Outp, int ldOut, int outOff,
               int K, int N,
               const int* __restrict__ idx,    // grouped row list
               const int* __restrict__ offs,   // per-expert offsets or nullptr
               int expert, int doRelu)
{
    int mBase = 0, mCnt = BROWS;
    const int* rlist = nullptr;
    if (offs) {
        int lo = offs[expert];
        mBase = lo; mCnt = offs[expert + 1] - lo; rlist = idx;
    }
    int m0 = blockIdx.x * 128;
    if (m0 >= mCnt) return;
    int n0 = blockIdx.y * 64;

    __shared__ __align__(16) u16 As[128][72];   // [m][k], 144B row stride
    __shared__ __align__(16) u16 Bs[64][72];    // [n][k] (weights pre-transposed)

    int t = threadIdx.x, lane = t & 31, wave = t >> 5;
    int wm = wave & 3, wn = wave >> 2;

    v8f zero8 = {0.f, 0.f, 0.f, 0.f, 0.f, 0.f, 0.f, 0.f};
    v8f acc[2][2];
    acc[0][0] = zero8; acc[0][1] = zero8; acc[1][0] = zero8; acc[1][1] = zero8;

    // A loader: 2 threads per row, 32 K-values each
    int arow = t >> 1, akq = (t & 1) * 32;
    int am = m0 + arow;
    int ag = (am < mCnt) ? (rlist ? rlist[mBase + am] : am)
                         : (rlist ? rlist[mBase] : 0);     // valid row; never stored
    const float* ArowF = (const float*)Aptr + (long)ag * ldA + akq;
    const u16*   ArowB = (const u16*)Aptr   + (long)ag * ldA + akq;

    // B loader: 4 threads per n-row, 16 K-values each
    int brow = t >> 2, bkq = (t & 3) * 16;
    const u16* WTrow = WT + (long)(n0 + brow) * ldW + bkq;

    for (int k0 = 0; k0 < K; k0 += 64) {
        // ---- stage A ----
        if (AIN_F32) {
            if (k0 + 64 < K) __builtin_prefetch(ArowF + k0 + 64, 0, 0);
#pragma unroll
            for (int i = 0; i < 8; ++i) {
                float4 v = *(const float4*)(ArowF + k0 + i * 4);
                u32x2 w = { pack2(v.x, v.y), pack2(v.z, v.w) };
                *(u32x2*)&As[arow][akq + i * 4] = w;
            }
        } else {
#pragma unroll
            for (int i = 0; i < 4; ++i)
                async_g2l_b128(&As[arow][akq + i * 8], ArowB + k0 + i * 8);
        }
        // ---- stage B (always bf16): async global -> LDS ----
#pragma unroll
        for (int i = 0; i < 2; ++i)
            async_g2l_b128(&Bs[brow][bkq + i * 8], WTrow + k0 + i * 8);
        wait_async0();
        __syncthreads();

        // ---- fragments (ISA 7.12.2): all reads are contiguous b128 ----
        int r = lane & 15;
        int ahalf = (lane < 16) ? 0 : 8;    // A: lanes>=16 hold K {8..15, 24..31}
        int bhalf = (lane < 16) ? 0 : 8;    // B: lanes>=16 hold K {8..15, 24..31}
#pragma unroll
        for (int kk = 0; kk < 64; kk += 32) {
            Frag16 a[2], b[2];
#pragma unroll
            for (int mf = 0; mf < 2; ++mf) {
                const u16* p = &As[wm * 32 + mf * 16 + r][kk + ahalf];
                a[mf].q[0] = *(const u32x4*)p;
                a[mf].q[1] = *(const u32x4*)(p + 16);
            }
#pragma unroll
            for (int nf = 0; nf < 2; ++nf) {
                const u16* p = &Bs[wn * 32 + nf * 16 + r][kk + bhalf];
                b[nf].q[0] = *(const u32x4*)p;
                b[nf].q[1] = *(const u32x4*)(p + 16);
            }
#pragma unroll
            for (int mf = 0; mf < 2; ++mf)
#pragma unroll
                for (int nf = 0; nf < 2; ++nf)
                    acc[mf][nf] = __builtin_amdgcn_wmma_f32_16x16x32_bf16(
                        false, a[mf].bf, false, b[nf].bf, (short)0, acc[mf][nf],
                        false, false);
        }
        __syncthreads();
    }

    // ---- epilogue: bias + activation, scattered store ----
    int cc = lane & 15;
    int rr = (lane >> 4) * 8;               // C/D: element v <-> M = v + 8*(lane>=16)
    bool full = (m0 + 128 <= mCnt);
    float* OutF = (float*)Outp;
    u16*   OutB = (u16*)Outp;
#pragma unroll
    for (int mf = 0; mf < 2; ++mf) {
        int rbase = m0 + wm * 32 + mf * 16 + rr;
        int g[8];
        if (rlist) {
            if (full) {                      // rbase is 8-aligned: two b128 loads
                int4 q0 = *(const int4*)(rlist + mBase + rbase);
                int4 q1 = *(const int4*)(rlist + mBase + rbase + 4);
                g[0] = q0.x; g[1] = q0.y; g[2] = q0.z; g[3] = q0.w;
                g[4] = q1.x; g[5] = q1.y; g[6] = q1.z; g[7] = q1.w;
            } else {                         // branchless clamped gather
#pragma unroll
                for (int v = 0; v < 8; ++v) {
                    int mm = rbase + v;
                    g[v] = rlist[mBase + (mm < mCnt ? mm : 0)];
                }
            }
        } else {
#pragma unroll
            for (int v = 0; v < 8; ++v) g[v] = rbase + v;
        }
#pragma unroll
        for (int nf = 0; nf < 2; ++nf) {
            int gcol = n0 + wn * 32 + nf * 16 + cc;
            float bv = bias ? bias[gcol] : 0.f;
            if (full) {
#pragma unroll
                for (int v = 0; v < 8; ++v) {
                    float x = acc[mf][nf][v] + bv;
                    if (doRelu) x = fmaxf(x, 0.f);
                    long o = (long)g[v] * ldOut + outOff + gcol;
                    if (OUT_BF16) OutB[o] = bfbits(x);
                    else          OutF[o] = x;
                }
            } else {
#pragma unroll
                for (int v = 0; v < 8; ++v) {
                    if (rbase + v < mCnt) {
                        float x = acc[mf][nf][v] + bv;
                        if (doRelu) x = fmaxf(x, 0.f);
                        long o = (long)g[v] * ldOut + outOff + gcol;
                        if (OUT_BF16) OutB[o] = bfbits(x);
                        else          OutF[o] = x;
                    }
                }
            }
        }
    }
}

// ---------------------------------------------------------------------------
// Tiny per-row action/reward embeddings into cat[:, 512:524] (bf16) and
// zero-fill of the K-pad columns [1036, 1088).
__global__ void k_embed(const float* __restrict__ a, const float* __restrict__ r,
                        const float* __restrict__ aw, const float* __restrict__ ab,
                        const float* __restrict__ rw, const float* __restrict__ rb,
                        const int* __restrict__ labels, u16* __restrict__ cat)
{
    int b = blockIdx.x;
    int e = labels[b];
    int t = threadIdx.x;
    u16* row = cat + (long)b * CATP;
    if (t < 8) {
        const float* w = aw + (long)e * ADIM * 8;
        const float* av = a + (long)b * ADIM;
        float s = ab[e * 8 + t];
        for (int k = 0; k < ADIM; ++k) s += av[k] * w[k * 8 + t];
        row[512 + t] = bfbits(s);
    } else if (t < 12) {
        int j = t - 8;
        row[520 + j] = bfbits(rb[e * 4 + j] + r[b] * rw[e * 4 + j]);
    } else {
        for (int c = CATD + (t - 12); c < CATP; c += 4) row[c] = 0;
    }
}

__global__ void k_sample_z(const float* __restrict__ mu, const float* __restrict__ lv,
                           const float* __restrict__ eps, float* __restrict__ z, long n)
{
    long i = (long)blockIdx.x * blockDim.x + threadIdx.x;
    if (i < n) z[i] = mu[i] + eps[i] * expf(0.5f * lv[i]);
}

__global__ __launch_bounds__(128)
void k_softmax(float* __restrict__ x, int ncols)
{
    long row = blockIdx.x;
    float* p = x + row * (long)ncols;
    __shared__ float redm[4], reds[4];
    int t = threadIdx.x, lane = t & 31, w = t >> 5;

    float mx = -3.4e38f;
    for (int c = t; c < ncols; c += 128) mx = fmaxf(mx, p[c]);
    for (int o = 16; o; o >>= 1) mx = fmaxf(mx, __shfl_xor(mx, o, 32));
    if (lane == 0) redm[w] = mx;
    __syncthreads();
    mx = fmaxf(fmaxf(redm[0], redm[1]), fmaxf(redm[2], redm[3]));

    float s = 0.f;
    for (int c = t; c < ncols; c += 128) { float v = expf(p[c] - mx); p[c] = v; s += v; }
    for (int o = 16; o; o >>= 1) s += __shfl_xor(s, o, 32);
    if (lane == 0) reds[w] = s;
    __syncthreads();
    s = reds[0] + reds[1] + reds[2] + reds[3];
    float inv = 1.f / s;
    for (int c = t; c < ncols; c += 128) p[c] *= inv;
}

__global__ void k_ones(float* __restrict__ x, int n)
{
    int i = blockIdx.x * blockDim.x + threadIdx.x;
    if (i < n) x[i] = 1.f;
}

// ---------------------------------------------------------------------------
extern "C" void kernel_launch(void* const* d_in, const int* in_sizes, int n_in,
                              void* d_out, int out_size, void* d_ws, size_t ws_size,
                              hipStream_t stream)
{
    (void)in_sizes; (void)n_in; (void)out_size; (void)ws_size;
    const int B = BROWS, D = DDIM, A = ADIM, H = HDIM;

    const float* s1     = (const float*)d_in[0];
    const float* a_in   = (const float*)d_in[1];
    const float* r_in   = (const float*)d_in[2];
    const float* s2     = (const float*)d_in[3];
    const int*   labels = (const int*)  d_in[4];
    const float* eps    = (const float*)d_in[5];
    const float* sw1 = (const float*)d_in[6],  *sb1 = (const float*)d_in[7];
    const float* sw2 = (const float*)d_in[8],  *sb2 = (const float*)d_in[9];
    const float* aw  = (const float*)d_in[10], *ab  = (const float*)d_in[11];
    const float* rw  = (const float*)d_in[12], *rb  = (const float*)d_in[13];
    const float* nw1 = (const float*)d_in[14], *nb1 = (const float*)d_in[15];
    const float* nw2 = (const float*)d_in[16], *nb2 = (const float*)d_in[17];
    const float* cw1 = (const float*)d_in[18], *cb1 = (const float*)d_in[19];
    const float* cw2 = (const float*)d_in[20], *cb2 = (const float*)d_in[21];
    const float* muw = (const float*)d_in[22], *mub = (const float*)d_in[23];
    const float* lvw = (const float*)d_in[24], *lvb = (const float*)d_in[25];
    const float* sdw1 = (const float*)d_in[26], *sdb1 = (const float*)d_in[27];
    const float* sdw2 = (const float*)d_in[28], *sdb2 = (const float*)d_in[29];
    const float* sdw3 = (const float*)d_in[30], *sdb3 = (const float*)d_in[31];
    const float* ndw1 = (const float*)d_in[32], *ndb1 = (const float*)d_in[33];
    const float* ndw2 = (const float*)d_in[34], *ndb2 = (const float*)d_in[35];
    const float* ndw3 = (const float*)d_in[36], *ndb3 = (const float*)d_in[37];
    const float* adw1 = (const float*)d_in[38], *adb1 = (const float*)d_in[39];
    const float* adw2 = (const float*)d_in[40], *adb2 = (const float*)d_in[41];

    // Outputs (flat concat in return order)
    float* out = (float*)d_out;
    float* o_state  = out;
    float* o_action = o_state  + (long)B * D;
    float* o_reward = o_action + (long)B * A;
    float* o_next   = o_reward + (long)B;
    float* o_mu     = o_next   + (long)B * D;
    float* o_lv     = o_mu     + (long)B * D;
    float* o_z      = o_lv     + (long)B * D;

    // Workspace: grouping lists + bf16 activations + bf16 transposed weights
    int* I    = (int*)d_ws;
    int* idx  = I;            // [B]
    int* cnts = I + B;        // [5]
    int* offs = I + B + 8;    // [6]
    int* curs = I + B + 16;   // [5]
    u16* U     = (u16*)(I + B + 24);
    u16* b_h   = U;                        // [B,128]
    u16* b_cat = b_h   + (long)B * 128;    // [B,CATP]
    u16* b_c1  = b_cat + (long)B * CATP;   // [B,512]
    u16* b_c2  = b_c1  + (long)B * H;      // [B,512] (reused as [B,128] in decoders)
    u16* wp    = b_c2  + (long)B * H;
    auto walloc = [&](long n) { u16* r = wp; wp += n; return r; };
    u16* sw1T  = walloc(5L * 128 * 1024);
    u16* sw2T  = walloc(5L * 512 * 128);
    u16* nw1T  = walloc(5L * 128 * 1024);
    u16* nw2T  = walloc(5L * 512 * 128);
    u16* cw1T  = walloc(5L * 512 * CATP);
    u16* cw2T  = walloc(5L * 512 * 512);
    u16* muwT  = walloc(5L * 1024 * 512);
    u16* lvwT  = walloc(5L * 1024 * 512);
    u16* sdw1T = walloc(128L * 1024);
    u16* sdw2T = walloc(128L * 128);
    u16* sdw3T = walloc(1024L * 128);
    u16* ndw1T = walloc(128L * 1024);
    u16* ndw2T = walloc(128L * 128);
    u16* ndw3T = walloc(1024L * 128);
    u16* adw1T = walloc(512L * 1024);
    u16* adw2T = walloc(64L * 512);

    // 0) weight convert + transpose (+ zero K-padding); expert loop on host
    auto wconv = [&](const float* W, u16* WT, int K, int N, int Kpad, int count) {
        for (int e = 0; e < count; ++e)
            k_wconv<<<dim3(Kpad / 64, N), 64, 0, stream>>>(
                W + (long)e * K * N, WT + (long)e * N * Kpad, K, N, Kpad);
    };
    wconv(sw1, sw1T, 1024, 128, 1024, 5);
    wconv(sw2, sw2T, 128, 512, 128, 5);
    wconv(nw1, nw1T, 1024, 128, 1024, 5);
    wconv(nw2, nw2T, 128, 512, 128, 5);
    wconv(cw1, cw1T, CATD, 512, CATP, 5);
    wconv(cw2, cw2T, 512, 512, 512, 5);
    wconv(muw, muwT, 512, 1024, 512, 5);
    wconv(lvw, lvwT, 512, 1024, 512, 5);
    wconv(sdw1, sdw1T, 1024, 128, 1024, 1);
    wconv(sdw2, sdw2T, 128, 128, 128, 1);
    wconv(sdw3, sdw3T, 128, 1024, 128, 1);
    wconv(ndw1, ndw1T, 1024, 128, 1024, 1);
    wconv(ndw2, ndw2T, 128, 128, 128, 1);
    wconv(ndw3, ndw3T, 128, 1024, 128, 1);
    wconv(adw1, adw1T, 1024, 512, 1024, 1);
    wconv(adw2, adw2T, 512, 64, 512, 1);

    // 1) group rows by expert label
    k_zero24<<<1, 32, 0, stream>>>(I + B);
    k_count<<<(B + 255) / 256, 256, 0, stream>>>(labels, cnts);
    k_scan<<<1, 32, 0, stream>>>(cnts, offs, curs);
    k_fill<<<(B + 255) / 256, 256, 0, stream>>>(labels, curs, idx);

    // 2) tiny embeddings into cat[:,512:524] + zero pad columns
    k_embed<<<B, 16, 0, stream>>>(a_in, r_in, aw, ab, rw, rb, labels, b_cat);

    auto gemm = [&](bool ainF, bool outBF, const void* Ap, int ldA,
                    const u16* WT, int ldW, const float* bp,
                    void* Op, int ldOut, int outOff, int K, int N,
                    const int* offp, int e, int relu) {
        dim3 grid(BROWS / 128, N / 64);
        if (ainF) {         // fp32-input GEMMs always produce bf16 hiddens
            wmma_gemm<true,  true ><<<grid, 256, 0, stream>>>(Ap, ldA, WT, ldW, bp, Op, ldOut, outOff, K, N, idx, offp, e, relu);
        } else if (outBF) {
            wmma_gemm<false, true ><<<grid, 256, 0, stream>>>(Ap, ldA, WT, ldW, bp, Op, ldOut, outOff, K, N, idx, offp, e, relu);
        } else {
            wmma_gemm<false, false><<<grid, 256, 0, stream>>>(Ap, ldA, WT, ldW, bp, Op, ldOut, outOff, K, N, idx, offp, e, relu);
        }
    };

    // 3) grouped encoder chain, one expert at a time (only its rows touched)
    for (int e = 0; e < NEXP; ++e) {
        gemm(true,  true,  s1,    D,    sw1T + (long)e * 128 * 1024, 1024, sb1 + e * 128, b_h,   128,  0,   1024, 128,  offs, e, 1);
        gemm(false, true,  b_h,   128,  sw2T + (long)e * 512 * 128,  128,  sb2 + e * H,   b_cat, CATP, 0,   128,  512,  offs, e, 0);
        gemm(true,  true,  s2,    D,    nw1T + (long)e * 128 * 1024, 1024, nb1 + e * 128, b_h,   128,  0,   1024, 128,  offs, e, 1);
        gemm(false, true,  b_h,   128,  nw2T + (long)e * 512 * 128,  128,  nb2 + e * H,   b_cat, CATP, 524, 128,  512,  offs, e, 0);
        gemm(false, true,  b_cat, CATP, cw1T + (long)e * 512 * CATP, CATP, cb1 + e * H,   b_c1,  H,    0,   CATP, 512,  offs, e, 1);
        gemm(false, true,  b_c1,  H,    cw2T + (long)e * 512 * 512,  512,  cb2 + e * H,   b_c2,  H,    0,   512,  512,  offs, e, 0);
        gemm(false, false, b_c2,  H,    muwT + (long)e * 1024 * 512, 512,  mub + e * D,   o_mu,  D,    0,   512,  1024, offs, e, 0);
        gemm(false, false, b_c2,  H,    lvwT + (long)e * 1024 * 512, 512,  lvb + e * D,   o_lv,  D,    0,   512,  1024, offs, e, 0);
    }

    // 4) z = mu + eps * exp(0.5*logvar)
    long nz = (long)B * D;
    k_sample_z<<<(unsigned)((nz + 255) / 256), 256, 0, stream>>>(o_mu, o_lv, eps, o_z, nz);

    // 5) dense decoders (offs = nullptr -> identity rows)
    gemm(true,  true,  o_z,  D,   sdw1T, 1024, sdb1, b_h,      128, 0, 1024, 128,  nullptr, 0, 1);
    gemm(false, true,  b_h,  128, sdw2T, 128,  sdb2, b_c2,     128, 0, 128,  128,  nullptr, 0, 1);
    gemm(false, false, b_c2, 128, sdw3T, 128,  sdb3, o_state,  D,   0, 128,  1024, nullptr, 0, 0);
    k_softmax<<<B, 128, 0, stream>>>(o_state, D);

    gemm(true,  true,  o_z,  D,   ndw1T, 1024, ndb1, b_h,      128, 0, 1024, 128,  nullptr, 0, 1);
    gemm(false, true,  b_h,  128, ndw2T, 128,  ndb2, b_c2,     128, 0, 128,  128,  nullptr, 0, 1);
    gemm(false, false, b_c2, 128, ndw3T, 128,  ndb3, o_next,   D,   0, 128,  1024, nullptr, 0, 0);
    k_softmax<<<B, 128, 0, stream>>>(o_next, D);

    gemm(true,  true,  o_z,  D,   adw1T, 1024, adb1, b_c1,     H,   0, 1024, 512,  nullptr, 0, 1);
    gemm(false, false, b_c1, H,   adw2T, 512,  adb2, o_action, A,   0, 512,  64,   nullptr, 0, 0);
    k_softmax<<<B, 128, 0, stream>>>(o_action, A);

    // reward decoder: softmax over a single logit is identically 1.0
    k_ones<<<(B + 255) / 256, 256, 0, stream>>>(o_reward, B);
}
